// JointWiseEncoder_12283606468127
// MI455X (gfx1250) — matrix operand order repaired and checked by reading, not compiled, not run on hardware
//
#include <hip/hip_runtime.h>
#include <hip/hip_bf16.h>

// Problem constants (from reference)
#define NB   8192
#define NT   64
#define NJ   17
#define ND   3
#define NH   128                 // H
#define NG   512                 // 4H
#define NSEQ (NB * NJ)           // 139264
#define NKO  (NJ * NH)           // 2176
#define NO   512                 // HIDDEN_DIM

typedef __attribute__((ext_vector_type(8)))  float   v8f;
typedef __attribute__((ext_vector_type(2)))  float   v2f;
typedef __attribute__((ext_vector_type(16))) __bf16  v16bf;

#if defined(__has_builtin)
#  if __has_builtin(__builtin_amdgcn_wmma_f32_16x16x4_f32)
#    define HAS_F32X4 1
#  else
#    define HAS_F32X4 0
#  endif
#else
#  define HAS_F32X4 0
#endif

union BfFrag { v16bf v; unsigned short u[16]; uint4 q[2]; };

__device__ __forceinline__ unsigned short f2bf_rne(float x) {
  unsigned int u = __float_as_uint(x);
  u += 0x7fffu + ((u >> 16) & 1u);
  return (unsigned short)(u >> 16);
}
__device__ __forceinline__ float fast_rcp(float x) { return __builtin_amdgcn_rcpf(x); }
// sigmoid via v_exp_f32 + v_rcp_f32 (no IEEE divide sequence)
__device__ __forceinline__ float sigm(float x) { return fast_rcp(1.0f + __expf(-x)); }
// tanh(x) = 2*sigmoid(2x) - 1
__device__ __forceinline__ float tanh_f(float x) {
  return fmaf(sigm(2.0f * x), 2.0f, -1.0f);
}

// ---------------------------------------------------------------------------
// Kernel 1: shared LSTM over N=B*J sequences, T=64 steps.
// Block: 512 threads = 16 waves, handles 16 sequences.
// Wave w owns 32 gate columns: gate = w>>2 (i,f,g,o), sub = w&3.
// gates[16x512] = [x_t|1]·[W_ih^T;bias]  (f32 WMMA 16x16x4)
//               + h·W_hh^T               (bf16 WMMA 16x16x32, fp32 accum)
// W_hh^T B-fragments are register-resident for the whole kernel.
// Activations are branchless: a = fma(sigmoid(inScale*v), outScale, outBias)
// with wave-uniform constants (tanh == 2*sigmoid(2v)-1 for the g gate).
// ---------------------------------------------------------------------------
__global__ __launch_bounds__(512)
void lstm_kernel(const float* __restrict__ pose, const float* __restrict__ W_ih,
                 const float* __restrict__ W_hh, const float* __restrict__ b_ih,
                 const float* __restrict__ b_hh, float* __restrict__ jf)
{
  __shared__ __attribute__((aligned(16))) unsigned short h_bf[16][136]; // padded rows (bank-conflict free b128)
  __shared__ __attribute__((aligned(16))) float x4[16][4];              // [x0,x1,x2,1]
  __shared__ float gbuf[16][516];                                       // activated gates, padded

  const int tid  = threadIdx.x;
  const int wave = tid >> 5;
  const int lane = tid & 31;
  const int lrow = lane & 15;
  const int lhi  = lane >> 4;
  const int gate = wave >> 2;
  const int sub  = wave & 3;
  const int colbase = 128 * gate + 32 * sub;
  const int seq0 = blockIdx.x * 16;

  // branchless activation constants (g-gate -> tanh, others -> sigmoid)
  const float aIn  = (gate == 2) ? -2.0f : -1.0f;   // folded negation for exp(-s*v)
  const float aOut = (gate == 2) ?  2.0f :  1.0f;
  const float aBias= (gate == 2) ? -1.0f :  0.0f;

  // --- register-resident W_hh^T B-fragments (bf16): 2 tiles x 4 K-chunks ---
  // B element e of lane l <-> K = k0 + 16*(l/16) + e, N = coltile + l%16
  v16bf Bw[2][4];
#pragma unroll
  for (int t2 = 0; t2 < 2; ++t2) {
    const int col = colbase + 16 * t2 + lrow;
#pragma unroll
    for (int c = 0; c < 4; ++c) {
      const float* wr = W_hh + col * NH + 32 * c + 16 * lhi;
      BfFrag f;
#pragma unroll
      for (int e = 0; e < 16; ++e) f.u[e] = f2bf_rne(wr[e]);
      Bw[t2][c] = f.v;
    }
  }

#if HAS_F32X4
  // B operand for the f32 16x16x4 init WMMA: rows k=0..2 = W_ih[col][k], k=3 = b_ih+b_hh
  v2f Bx[2];
#pragma unroll
  for (int t2 = 0; t2 < 2; ++t2) {
    const int col = colbase + 16 * t2 + lrow;
    const float bsum = b_ih[col] + b_hh[col];
    v2f b;
    if (lhi == 0) { b.x = W_ih[col * 3 + 0]; b.y = W_ih[col * 3 + 1]; }
    else          { b.x = W_ih[col * 3 + 2]; b.y = bsum; }
    Bx[t2] = b;
  }
#else
  float wA[2], wB[2], wC[2], bs[2];
#pragma unroll
  for (int t2 = 0; t2 < 2; ++t2) {
    const int col = colbase + 16 * t2 + lrow;
    wA[t2] = W_ih[col * 3 + 0]; wB[t2] = W_ih[col * 3 + 1]; wC[t2] = W_ih[col * 3 + 2];
    bs[t2] = b_ih[col] + b_hh[col];
  }
#endif

  // c/h ownership: wave owns 8 hidden cols; lane owns 4 (row,col) cells
  const int hcol = 8 * wave + (lane & 7);
  const int rb   = lane >> 3;
  float cst[4] = {0.f, 0.f, 0.f, 0.f};
  float hq[4]  = {0.f, 0.f, 0.f, 0.f};

  for (int i = tid; i < 16 * 136; i += 512) ((unsigned short*)h_bf)[i] = 0;
  if (tid < 16) x4[tid][3] = 1.0f;

  // x staging: seq n = seq0+r, x[n,t,:] = pose[b, t, j, :] with b=n/17, j=n%17
  const int xr = tid / 3, xd = tid % 3;
  long xbase = 0;
  if (tid < 48) {
    const int n  = seq0 + xr;
    const int bb = n / NJ, jj = n % NJ;
    xbase = (long)(bb * NT) * (NJ * ND) + (long)jj * ND + xd;
  }

  for (int t = 0; t < NT; ++t) {
    if (tid < 48) x4[xr][xd] = pose[xbase + (long)t * (NJ * ND)];
    __syncthreads();   // x_t ready; previous step's h_bf writes & gbuf reads done

    v8f acc[2];
#if HAS_F32X4
    {
      v2f Ax;  // A 16x4 f32: lane row = l%16, K = 2*(l/16)+{0,1}
      Ax.x = x4[lrow][2 * lhi];
      Ax.y = x4[lrow][2 * lhi + 1];
      v8f zz = {0.f, 0.f, 0.f, 0.f, 0.f, 0.f, 0.f, 0.f};
#pragma unroll
      for (int t2 = 0; t2 < 2; ++t2)
        acc[t2] = __builtin_amdgcn_wmma_f32_16x16x4_f32(false, Ax, false, Bx[t2],
                                                        (short)0, zz, false, false);
    }
#else
#pragma unroll
    for (int t2 = 0; t2 < 2; ++t2) {
#pragma unroll
      for (int e = 0; e < 8; ++e) {
        const int r = e + 8 * lhi;
        acc[t2][e] = bs[t2] + wA[t2] * x4[r][0] + wB[t2] * x4[r][1] + wC[t2] * x4[r][2];
      }
    }
#endif

    // A fragments (bf16 16x32): element e <-> K = k0 + 16*(l/16) + e
#pragma unroll
    for (int c = 0; c < 4; ++c) {
      BfFrag a;
      a.q[0] = *(const uint4*)&h_bf[lrow][32 * c + 16 * lhi];
      a.q[1] = *(const uint4*)&h_bf[lrow][32 * c + 16 * lhi + 8];
      acc[0] = __builtin_amdgcn_wmma_f32_16x16x32_bf16(false, a.v, false, Bw[0][c],
                                                       (short)0, acc[0], false, false);
      acc[1] = __builtin_amdgcn_wmma_f32_16x16x32_bf16(false, a.v, false, Bw[1][c],
                                                       (short)0, acc[1], false, false);
    }

    // branchless activation, scatter to gbuf
#pragma unroll
    for (int t2 = 0; t2 < 2; ++t2) {
#pragma unroll
      for (int e = 0; e < 8; ++e) {
        const float v = acc[t2][e];   // C layout: row = e + 8*(l/16), col = tile + l%16
        const float s = fast_rcp(1.0f + __expf(aIn * v));
        const float a = fmaf(s, aOut, aBias);
        gbuf[e + 8 * lhi][colbase + 16 * t2 + lrow] = a;
      }
    }
    __syncthreads();   // gbuf complete; all h_bf reads of this step done

    // c/h update for owned cells; write h back as bf16 for next step
#pragma unroll
    for (int q = 0; q < 4; ++q) {
      const int r = rb + 4 * q;
      const float iv = gbuf[r][hcol];
      const float fv = gbuf[r][128 + hcol];
      const float gv = gbuf[r][256 + hcol];
      const float ov = gbuf[r][384 + hcol];
      const float cc = fv * cst[q] + iv * gv;
      cst[q] = cc;
      // tanh(cc) = 2*sigmoid(2cc)-1, rcp-based
      const float th = fmaf(fast_rcp(1.0f + __expf(-2.0f * cc)), 2.0f, -1.0f);
      const float hh = ov * th;
      hq[q] = hh;
      h_bf[r][hcol] = f2bf_rne(hh);
    }
  }

#pragma unroll
  for (int q = 0; q < 4; ++q) {
    const int r = rb + 4 * q;
    jf[(size_t)(seq0 + r) * NH + hcol] = hq[q];   // h_last, fp32
  }
}

// ---------------------------------------------------------------------------
// Kernel 2: per-batch-element MHA over J=17 joints + out_proj (memory-bound,
// VALU fp32). Writes attended in place over jf.
// ---------------------------------------------------------------------------
__global__ __launch_bounds__(128)
void attn_kernel(const float* __restrict__ in_w, const float* __restrict__ in_b,
                 const float* __restrict__ out_w, const float* __restrict__ out_b,
                 float* __restrict__ jf)
{
  __shared__ float jfL[NJ][NH];
  __shared__ float qkvL[NJ][3 * NH];
  __shared__ float ctxL[NJ][NH];
  const int tid = threadIdx.x;
  const size_t base = (size_t)blockIdx.x * NKO;

  for (int i = tid; i < NJ * NH; i += 128) jfL[i / NH][i % NH] = jf[base + i];
  __syncthreads();

#pragma unroll
  for (int cc = 0; cc < 3; ++cc) {               // qkv: 3 cols per thread
    const int col = tid + 128 * cc;
    const float* w = in_w + (size_t)col * NH;
    float acc[NJ];
#pragma unroll
    for (int j = 0; j < NJ; ++j) acc[j] = 0.f;
    for (int k = 0; k < NH; ++k) {
      const float wk = w[k];
#pragma unroll
      for (int j = 0; j < NJ; ++j) acc[j] += jfL[j][k] * wk;
    }
    const float bb = in_b[col];
#pragma unroll
    for (int j = 0; j < NJ; ++j) qkvL[j][col] = acc[j] + bb;
  }
  __syncthreads();

  if (tid < 2 * NJ) {                            // one (head, query) per thread
    const int hh = tid / NJ, qq = tid % NJ;
    const int qo = hh * 64, ko = NH + hh * 64, vo = 2 * NH + hh * 64;
    float sc[NJ];
    float mx = -3.0e38f;
    for (int kk = 0; kk < NJ; ++kk) {
      float s = 0.f;
      for (int e = 0; e < 64; ++e) s += qkvL[qq][qo + e] * qkvL[kk][ko + e];
      s *= 0.125f;                               // 1/sqrt(64)
      sc[kk] = s;
      mx = fmaxf(mx, s);
    }
    float sum = 0.f;
    for (int kk = 0; kk < NJ; ++kk) { sc[kk] = __expf(sc[kk] - mx); sum += sc[kk]; }
    const float inv = fast_rcp(sum);
    for (int e = 0; e < 64; ++e) {
      float c = 0.f;
      for (int kk = 0; kk < NJ; ++kk) c += sc[kk] * qkvL[kk][vo + e];
      ctxL[qq][hh * 64 + e] = c * inv;
    }
  }
  __syncthreads();

  {                                              // out_proj: 1 col per thread
    const int col = tid;
    const float* w = out_w + (size_t)col * NH;
    float acc[NJ];
#pragma unroll
    for (int j = 0; j < NJ; ++j) acc[j] = 0.f;
    for (int k = 0; k < NH; ++k) {
      const float wk = w[k];
#pragma unroll
      for (int j = 0; j < NJ; ++j) acc[j] += ctxL[j][k] * wk;
    }
    const float bb = out_b[col];
#pragma unroll
    for (int j = 0; j < NJ; ++j) jf[base + (size_t)j * NH + col] = acc[j] + bb;
  }
}

// ---------------------------------------------------------------------------
// Kernel 3: out[8192,512] = attended[8192,2176] @ proj_w^T + proj_b
// fp32-exact GEMM with V_WMMA_F32_16X16X4_F32, LDS-staged A/B tiles.
// Block: 128 threads (4 waves) -> 16 rows x 512 cols.
// ---------------------------------------------------------------------------
#define KC 16
__global__ __launch_bounds__(128)
void proj_kernel(const float* __restrict__ att, const float* __restrict__ proj_w,
                 const float* __restrict__ proj_b, float* __restrict__ out)
{
  __shared__ __attribute__((aligned(16))) float aT[16][KC];
  __shared__ __attribute__((aligned(16))) float bT[NO][KC];
  const int tid  = threadIdx.x;
  const int wave = tid >> 5, lane = tid & 31, lrow = lane & 15, lhi = lane >> 4;
  const int row0 = blockIdx.x * 16;

  int cols[8];
  v8f acc[8];
#pragma unroll
  for (int j = 0; j < 8; ++j) {
    cols[j] = 128 * wave + 16 * j + lrow;
    const float b = proj_b[cols[j]];
#pragma unroll
    for (int e = 0; e < 8; ++e) acc[j][e] = b;   // bias: same col across all 8 rows
  }

  for (int kc = 0; kc < NKO / KC; ++kc) {        // 136 K-chunks
    for (int i = tid; i < 16 * KC; i += 128) {
      const int r = i / KC, k = i % KC;
      aT[r][k] = att[(size_t)(row0 + r) * NKO + kc * KC + k];
    }
    for (int i = tid; i < NO * (KC / 4); i += 128) {
      const int col = i / (KC / 4), k4 = i % (KC / 4);
      *(float4*)&bT[col][k4 * 4] =
          *(const float4*)&proj_w[(size_t)col * NKO + kc * KC + k4 * 4];
    }
    __syncthreads();
#if HAS_F32X4
#pragma unroll
    for (int cc = 0; cc < KC / 4; ++cc) {
      v2f Ax;
      Ax.x = aT[lrow][cc * 4 + 2 * lhi];
      Ax.y = aT[lrow][cc * 4 + 2 * lhi + 1];
#pragma unroll
      for (int j = 0; j < 8; ++j) {
        v2f Bxx;
        Bxx.x = bT[cols[j]][cc * 4 + 2 * lhi];
        Bxx.y = bT[cols[j]][cc * 4 + 2 * lhi + 1];
        acc[j] = __builtin_amdgcn_wmma_f32_16x16x4_f32(false, Ax, false, Bxx,
                                                       (short)0, acc[j], false, false);
      }
    }
#else
#pragma unroll
    for (int e = 0; e < 8; ++e) {
      const int r = e + 8 * lhi;
#pragma unroll
      for (int j = 0; j < 8; ++j) {
        float s = acc[j][e];
        for (int k = 0; k < KC; ++k) s += aT[r][k] * bT[cols[j]][k];
        acc[j][e] = s;
      }
    }
#endif
    __syncthreads();
  }

#pragma unroll
  for (int j = 0; j < 8; ++j)
#pragma unroll
    for (int e = 0; e < 8; ++e)
      out[(size_t)(row0 + e + 8 * lhi) * NO + cols[j]] = acc[j][e];
}

// ---------------------------------------------------------------------------
extern "C" void kernel_launch(void* const* d_in, const int* in_sizes, int n_in,
                              void* d_out, int out_size, void* d_ws, size_t ws_size,
                              hipStream_t stream)
{
  (void)in_sizes; (void)n_in; (void)out_size; (void)ws_size;
  const float* pose = (const float*)d_in[0];
  const float* W_ih = (const float*)d_in[1];
  const float* W_hh = (const float*)d_in[2];
  const float* b_ih = (const float*)d_in[3];
  const float* b_hh = (const float*)d_in[4];
  const float* ipw  = (const float*)d_in[5];
  const float* ipb  = (const float*)d_in[6];
  const float* opw  = (const float*)d_in[7];
  const float* opb  = (const float*)d_in[8];
  const float* pw   = (const float*)d_in[9];
  const float* pb   = (const float*)d_in[10];
  float* out = (float*)d_out;
  float* jf  = (float*)d_ws;   // NSEQ*NH fp32 = 71.3 MB scratch (jf -> attended in place)

  lstm_kernel<<<NSEQ / 16, 512, 0, stream>>>(pose, W_ih, W_hh, b_ih, b_hh, jf);
  attn_kernel<<<NB, 128, 0, stream>>>(ipw, ipb, opw, opb, jf);
  proj_kernel<<<NB / 16, 128, 0, stream>>>(jf, pw, pb, out);
}
